// MultiModalAttention_29729763623215
// MI455X (gfx1250) — matrix operand order repaired and checked by reading, not compiled, not run on hardware
//
#include <hip/hip_runtime.h>
#include <hip/hip_bf16.h>
#include <stdint.h>

#define H      16
#define DM     1024
#define MODAL  3
#define DK     64
#define BB     16
#define SS     1024
#define NROWS  (BB*SS)        // 16384
#define LDX    (MODAL*DM)     // 3072

typedef __attribute__((ext_vector_type(16))) __bf16         v16bf;
typedef __attribute__((ext_vector_type(8)))  float          v8f;
typedef __attribute__((ext_vector_type(4)))  float          v4f;
typedef __attribute__((ext_vector_type(8)))  unsigned short v8u16;
typedef __attribute__((ext_vector_type(16))) unsigned short v16u16;

union Frag {
    v16bf  bf;
    v16u16 u;
    v8u16  u8[2];
};

__device__ __forceinline__ unsigned short f2bfu(float f) {
    union { float f; unsigned u; } v; v.f = f;
    unsigned u = v.u;
    u += 0x7fffu + ((u >> 16) & 1u);           // round-to-nearest-even
    return (unsigned short)(u >> 16);
}
__device__ __forceinline__ float bf2f(unsigned short s) {
    union { unsigned u; float f; } v; v.u = ((unsigned)s) << 16;
    return v.f;
}

// ---------------------------------------------------------------------------
// Kernel 0: weight convert + transpose.  W[t][m][k][e] f32  ->  Wt[t*3+m][e][k] bf16
// ---------------------------------------------------------------------------
__global__ void wconvert_kernel(const float* __restrict__ Wq, const float* __restrict__ Wk,
                                const float* __restrict__ Wv, const float* __restrict__ Wo,
                                unsigned short* __restrict__ Wt) {
    size_t i = (size_t)blockIdx.x * 256 + threadIdx.x;   // 12*1024*1024 total
    int k  = (int)(i & (DM - 1));
    size_t r = i >> 10;
    int e  = (int)(r & (DM - 1));
    int tm = (int)(r >> 10);                  // 0..11 == t*3+m
    int t  = tm / 3;
    int m  = tm - t * 3;
    const float* W = (t == 0) ? Wq : (t == 1) ? Wk : (t == 2) ? Wv : Wo;
    Wt[i] = f2bfu(W[(size_t)m * DM * DM + (size_t)k * DM + e]);
}

// ---------------------------------------------------------------------------
// Fragment loaders (ISA 16x16x32 bf16 layouts)
//   A 16x32: element j -> k = (j>>3)*16 + lh*8 + (j&7), row = ll
//   B 32x16: element j -> k = (j>>3)*16 + lh*8 + (j&7), col = ll
// ---------------------------------------------------------------------------
__device__ __forceinline__ Frag load_b_frag(const unsigned short* __restrict__ bp) {
    Frag b;
    b.u8[0] = *(const v8u16*)(bp);
    b.u8[1] = *(const v8u16*)(bp + 16);
    return b;
}

__device__ __forceinline__ Frag load_a_frag_f32(const float* __restrict__ ap) {
    v4f a0 = *(const v4f*)(ap);
    v4f a1 = *(const v4f*)(ap + 4);
    v4f a2 = *(const v4f*)(ap + 16);
    v4f a3 = *(const v4f*)(ap + 20);
    Frag a;
#pragma unroll
    for (int j = 0; j < 4; j++) {
        a.bf[j]      = (__bf16)a0[j];
        a.bf[4 + j]  = (__bf16)a1[j];
        a.bf[8 + j]  = (__bf16)a2[j];
        a.bf[12 + j] = (__bf16)a3[j];
    }
    return a;
}

__device__ __forceinline__ Frag load_a_frag_bf16(const unsigned short* __restrict__ ap) {
    Frag a;
    a.u8[0] = *(const v8u16*)(ap);
    a.u8[1] = *(const v8u16*)(ap + 16);
    return a;
}

// ---------------------------------------------------------------------------
// WMMA GEMM core: C[16384 x 1024] = A[:, m*1024 : m*1024+1024] * Wt^T + bias
//   A row stride = 3072 (f32 or bf16), Wt is [1024 e][1024 k] bf16 (K-contig).
//   Block = 256 thr = 8 waves; wave -> 16 rows x 128 cols (8 wmma tiles).
//   B fragments ping-pong between two 8-frag register buffers: prefetch for
//   K-step i+1 writes the buffer not being consumed, so no register copies
//   and no loadcnt==0 stalls in front of the WMMAs.
// ---------------------------------------------------------------------------
template <bool AF32, bool OBF16>
__device__ __forceinline__ void gemm_core(const void* __restrict__ A_, int m,
                                          const unsigned short* __restrict__ W,
                                          const float* __restrict__ bias,
                                          void* __restrict__ Out,
                                          int bx, int by, int tid) {
    const int lane    = tid & 31;
    const int wave    = tid >> 5;
    const int rowBase = by * 128 + wave * 16;
    const int colBase = bx * 128;
    const int lh      = lane >> 4;   // 0 or 1 (half-wave)
    const int ll      = lane & 15;

    // Per-lane base pointers
    const float*          apF = AF32 ? (const float*)A_ +
        (size_t)(rowBase + ll) * LDX + (size_t)m * DM + lh * 8 : nullptr;
    const unsigned short* apB = !AF32 ? (const unsigned short*)A_ +
        (size_t)(rowBase + ll) * LDX + (size_t)m * DM + lh * 8 : nullptr;
    const unsigned short* wp  = W + (size_t)(colBase + ll) * DM + lh * 8;

    v8f acc[8] = {};

    // ---- prime: B fragments for kb = 0 into buffer 0
    Frag b0[8], b1[8];
#pragma unroll
    for (int t = 0; t < 8; t++) b0[t] = load_b_frag(wp + (size_t)(t * 16) * DM);

    // ---- main loop, 2 K-steps per trip (ping-pong), last 2 steps peeled
#pragma unroll 1
    for (int kb = 0; kb < DM - 64; kb += 64) {
        Frag a0 = AF32 ? load_a_frag_f32(apF + kb) : load_a_frag_bf16(apB + kb);
#pragma unroll
        for (int t = 0; t < 8; t++) {
            b1[t] = load_b_frag(wp + (size_t)(t * 16) * DM + kb + 32);
            acc[t] = __builtin_amdgcn_wmma_f32_16x16x32_bf16(
                false, a0.bf, false, b0[t].bf, (short)0, acc[t], false, false);
        }
        Frag a1 = AF32 ? load_a_frag_f32(apF + kb + 32) : load_a_frag_bf16(apB + kb + 32);
#pragma unroll
        for (int t = 0; t < 8; t++) {
            b0[t] = load_b_frag(wp + (size_t)(t * 16) * DM + kb + 64);
            acc[t] = __builtin_amdgcn_wmma_f32_16x16x32_bf16(
                false, a1.bf, false, b1[t].bf, (short)0, acc[t], false, false);
        }
    }
    {   // tail: kb = DM-64 and DM-32
        Frag a0 = AF32 ? load_a_frag_f32(apF + DM - 64) : load_a_frag_bf16(apB + DM - 64);
#pragma unroll
        for (int t = 0; t < 8; t++) {
            b1[t] = load_b_frag(wp + (size_t)(t * 16) * DM + DM - 32);
            acc[t] = __builtin_amdgcn_wmma_f32_16x16x32_bf16(
                false, a0.bf, false, b0[t].bf, (short)0, acc[t], false, false);
        }
        Frag a1 = AF32 ? load_a_frag_f32(apF + DM - 32) : load_a_frag_bf16(apB + DM - 32);
#pragma unroll
        for (int t = 0; t < 8; t++)
            acc[t] = __builtin_amdgcn_wmma_f32_16x16x32_bf16(
                false, a1.bf, false, b1[t].bf, (short)0, acc[t], false, false);
    }

    // ---- epilogue: D vgpr g -> row = rowBase + g + lh*8, col = colBase + t*16 + ll
#pragma unroll
    for (int t = 0; t < 8; t++) {
        int e   = colBase + t * 16 + ll;
        float bv = bias[e];
#pragma unroll
        for (int g = 0; g < 8; g++) {
            int r = rowBase + g + lh * 8;
            size_t off = (size_t)r * LDX + (size_t)m * DM + e;
            float val = acc[t][g] + bv;
            if (OBF16) ((unsigned short*)Out)[off] = f2bfu(val);
            else       ((float*)Out)[off]          = val;
        }
    }
}

// grid = (8, 128, 9)  z = p*3 + m,  p in {q,k,v}
__global__ void __launch_bounds__(256) qkv_gemm_kernel(
        const float* __restrict__ q, const float* __restrict__ k,
        const float* __restrict__ v,
        const unsigned short* __restrict__ Wt,     // [12][1024][1024] bf16
        const float* __restrict__ bq, const float* __restrict__ bk,
        const float* __restrict__ bv_,
        unsigned short* __restrict__ qkvb) {       // [3][16384][3072] bf16
    int z = blockIdx.z;
    int p = z / 3;
    int m = z - p * 3;
    const float* A    = (p == 0) ? q : (p == 1) ? k : v;
    const float* bias = ((p == 0) ? bq : (p == 1) ? bk : bv_) + m * DM;
    gemm_core<true, true>(A, m, Wt + (size_t)z * DM * DM, bias,
                          qkvb + (size_t)p * NROWS * LDX,
                          blockIdx.x, blockIdx.y, threadIdx.x);
}

// grid = (8, 128, 3)  z = m
__global__ void __launch_bounds__(256) out_gemm_kernel(
        const unsigned short* __restrict__ xb,
        const unsigned short* __restrict__ Wt,
        const float* __restrict__ bo, float* __restrict__ out) {
    int m = blockIdx.z;
    gemm_core<false, false>(xb, m, Wt + (size_t)(9 + m) * DM * DM, bo + m * DM,
                            out, blockIdx.x, blockIdx.y, threadIdx.x);
}

// ---------------------------------------------------------------------------
// Modal attention: per (n,h): 3x3 scores over DK=64, softmax over M=3, x = p*v
// ---------------------------------------------------------------------------
__global__ void modal_attn_kernel(const unsigned short* __restrict__ qb,
                                  const unsigned short* __restrict__ kb_,
                                  const unsigned short* __restrict__ vb,
                                  unsigned short* __restrict__ xb) {
    int tid = blockIdx.x * 256 + threadIdx.x;     // 0 .. 262143
    int h = tid & (H - 1);
    int n = tid >> 4;
    size_t base = (size_t)n * LDX + (size_t)h * DK;

    float s[3][3];
#pragma unroll
    for (int mi = 0; mi < 3; mi++)
#pragma unroll
        for (int ni = 0; ni < 3; ni++) s[mi][ni] = 0.f;

    for (int dc = 0; dc < DK; dc += 8) {
        float qv[3][8], kv[3][8];
#pragma unroll
        for (int mi = 0; mi < 3; mi++) {
            v8u16 qc = *(const v8u16*)(qb + base + mi * DM + dc);
            v8u16 kc = *(const v8u16*)(kb_ + base + mi * DM + dc);
#pragma unroll
            for (int j = 0; j < 8; j++) { qv[mi][j] = bf2f(qc[j]); kv[mi][j] = bf2f(kc[j]); }
        }
#pragma unroll
        for (int mi = 0; mi < 3; mi++)
#pragma unroll
            for (int ni = 0; ni < 3; ni++) {
                float acc = s[mi][ni];
#pragma unroll
                for (int j = 0; j < 8; j++) acc += qv[mi][j] * kv[ni][j];
                s[mi][ni] = acc;
            }
    }

    float p[3][3];
#pragma unroll
    for (int mi = 0; mi < 3; mi++) {
        float s0 = s[mi][0] * 0.125f;   // 1/sqrt(64)
        float s1 = s[mi][1] * 0.125f;
        float s2 = s[mi][2] * 0.125f;
        float mx = fmaxf(fmaxf(s0, s1), s2);
        float e0 = __expf(s0 - mx);
        float e1 = __expf(s1 - mx);
        float e2 = __expf(s2 - mx);
        float inv = 1.f / (e0 + e1 + e2);
        p[mi][0] = e0 * inv; p[mi][1] = e1 * inv; p[mi][2] = e2 * inv;
    }

    for (int dc = 0; dc < DK; dc += 8) {
        float vv[3][8];
#pragma unroll
        for (int ni = 0; ni < 3; ni++) {
            v8u16 vc = *(const v8u16*)(vb + base + ni * DM + dc);
#pragma unroll
            for (int j = 0; j < 8; j++) vv[ni][j] = bf2f(vc[j]);
        }
#pragma unroll
        for (int mi = 0; mi < 3; mi++) {
            v8u16 ov;
#pragma unroll
            for (int j = 0; j < 8; j++) {
                float x = p[mi][0] * vv[0][j] + p[mi][1] * vv[1][j] + p[mi][2] * vv[2][j];
                ov[j] = f2bfu(x);
            }
            *(v8u16*)(xb + base + mi * DM + dc) = ov;
        }
    }
}

// ---------------------------------------------------------------------------
extern "C" void kernel_launch(void* const* d_in, const int* in_sizes, int n_in,
                              void* d_out, int out_size, void* d_ws, size_t ws_size,
                              hipStream_t stream) {
    const float* q  = (const float*)d_in[0];
    const float* k  = (const float*)d_in[1];
    const float* v  = (const float*)d_in[2];
    const float* Wq = (const float*)d_in[3];
    const float* bq = (const float*)d_in[4];
    const float* Wk = (const float*)d_in[5];
    const float* bk = (const float*)d_in[6];
    const float* Wv = (const float*)d_in[7];
    const float* bv = (const float*)d_in[8];
    const float* Wo = (const float*)d_in[9];
    const float* bo = (const float*)d_in[10];
    float* out = (float*)d_out;

    unsigned char* ws = (unsigned char*)d_ws;
    unsigned short* Wt   = (unsigned short*)ws;                               // 24 MiB
    unsigned short* qkvb = (unsigned short*)(ws + (size_t)12 * DM * DM * 2);  // 288 MiB
    unsigned short* xb   = qkvb + (size_t)3 * NROWS * LDX;                    // 96 MiB

    hipLaunchKernelGGL(wconvert_kernel, dim3((12u * DM * DM) / 256), dim3(256), 0, stream,
                       Wq, Wk, Wv, Wo, Wt);
    hipLaunchKernelGGL(qkv_gemm_kernel, dim3(DM / 128, NROWS / 128, 9), dim3(256), 0, stream,
                       q, k, v, Wt, bq, bk, bv, qkvb);
    hipLaunchKernelGGL(modal_attn_kernel, dim3((NROWS * H) / 256), dim3(256), 0, stream,
                       qkvb, qkvb + (size_t)NROWS * LDX, qkvb + (size_t)2 * NROWS * LDX, xb);
    hipLaunchKernelGGL(out_gemm_kernel, dim3(DM / 128, NROWS / 128, 3), dim3(256), 0, stream,
                       xb, Wt, bo, out);
}
